// ConvTransformer_34265249087837
// MI455X (gfx1250) — compile-verified, matrix operand order checked by reading, not run on hardware
//
#include <hip/hip_runtime.h>
#include <hip/hip_bf16.h>

// ---------------------------------------------------------------------------
// Types for CDNA5 WMMA (wave32, 16x16x32 bf16 -> f32)
// ---------------------------------------------------------------------------
typedef __bf16 bf16;
typedef __attribute__((ext_vector_type(16))) __bf16 v16bf;
typedef __attribute__((ext_vector_type(8)))  __bf16 v8bf;
typedef __attribute__((ext_vector_type(8)))  float  v8f;

__device__ inline unsigned short f2bfbits(float f) {
  union { float f; unsigned u; } a; a.f = f;
  unsigned u = a.u + 0x7FFFu + ((a.u >> 16) & 1u);   // round-to-nearest-even
  return (unsigned short)(u >> 16);
}
__device__ inline float bf2f(unsigned short b) {
  union { unsigned u; float f; } a; a.u = (unsigned)b << 16; return a.f;
}

// ---------------------------------------------------------------------------
// Async global->LDS support (CDNA5).  Probe result (round 3 diagnostic):
//   __builtin_amdgcn_global_load_async_to_lds_b128 exists, 4 args,
//   parameter type = generic `int __attribute__((vector_size(16)))*`
//   (clang inserts the addrspace casts at codegen).
// Guarded: if the builtins are absent we fall back to plain vector loads.
// ---------------------------------------------------------------------------
#if __has_builtin(__builtin_amdgcn_global_load_async_to_lds_b128) && \
    __has_builtin(__builtin_amdgcn_s_wait_asynccnt)
#define ATTN_ASYNC 1
typedef int v4i_vs __attribute__((vector_size(16)));
__device__ inline void async_ld16(const void* g, void* l) {
  __builtin_amdgcn_global_load_async_to_lds_b128(
      (v4i_vs*)g, (v4i_vs*)l, 0, 0);
}
#else
#define ATTN_ASYNC 0
#endif

// ---------------------------------------------------------------------------
// Generic implicit-GEMM:  C[M,N] = A[M,K] * B[K,N]   (bf16 inputs, f32 acc)
// Block: 256 threads = 8 wave32.  Tile: BM=MT*16, BN=256, BK=32.
// Wave w computes the BMx32 strip at columns [w*32, w*32+32) (2 N-substrips),
// so each A fragment feeds 2 WMMAs.  MT=2 for M=32 layers (no wasted MACs),
// MT=4 for larger M.
// LDS: As[BM][32] (M x K), BsT[256][32] (N x K, i.e. B transposed).
// Fragment layouts follow CDNA5 ISA 7.12.2:
//   A (16x32 bf16): lane<16  row=lane,   elems = K[0..7],K[16..23]
//                   lane>=16 row=lane-16,elems = K[8..15],K[24..31]
//   B (32x16 bf16): same indexing with N in place of M.
//   C/D (16x16 f32): lane<16 -> N=lane, M=r ; lane>=16 -> N=lane-16, M=r+8.
// ---------------------------------------------------------------------------
template <int MT, class AGet, class BGet, class CPut>
__global__ void __launch_bounds__(256)
gemm_bf16(AGet aget, BGet bget, CPut cput, int M, int N, int K) {
  __shared__ __align__(16) bf16 As[MT * 16][32];
  __shared__ __align__(16) bf16 BsT[256][32];

  const int nbase = blockIdx.x * 256;
  const int mbase = blockIdx.y * (MT * 16);
  const int batch = blockIdx.z;

  const int tid  = threadIdx.x;
  const int wave = tid >> 5;
  const int lane = tid & 31;
  const int half = lane >> 4;   // 0 or 1
  const int l15  = lane & 15;

  v8f acc[MT][2];
  const v8f vz = {0.f,0.f,0.f,0.f,0.f,0.f,0.f,0.f};
#pragma unroll
  for (int i = 0; i < MT; ++i) { acc[i][0] = vz; acc[i][1] = vz; }

  for (int k0 = 0; k0 < K; k0 += 32) {
    // ---- stage A tile (BMx32), packed 2xbf16 dword stores ----
    for (int i = tid; i < MT * 16 * 16; i += 256) {
      int r = i >> 4, kk = (i & 15) * 2;
      float v0 = 0.f, v1 = 0.f;
      if (mbase + r < M) {
        if (k0 + kk     < K) v0 = aget(batch, mbase + r, k0 + kk);
        if (k0 + kk + 1 < K) v1 = aget(batch, mbase + r, k0 + kk + 1);
      }
      unsigned pk = (unsigned)f2bfbits(v0) | ((unsigned)f2bfbits(v1) << 16);
      *(unsigned*)&As[r][kk] = pk;
    }
    // ---- stage B tile transposed (256 x 32), packed dword stores ----
    for (int i = tid; i < 256 * 16; i += 256) {
      int c = i >> 4, kk = (i & 15) * 2;
      float v0 = 0.f, v1 = 0.f;
      if (nbase + c < N) {
        if (k0 + kk     < K) v0 = bget(batch, k0 + kk,     nbase + c);
        if (k0 + kk + 1 < K) v1 = bget(batch, k0 + kk + 1, nbase + c);
      }
      unsigned pk = (unsigned)f2bfbits(v0) | ((unsigned)f2bfbits(v1) << 16);
      *(unsigned*)&BsT[c][kk] = pk;
    }
    __syncthreads();

    // ---- two B fragments for this wave's 32-column strip ----
    const int col0 = wave * 32 + l15;
    v8bf b0lo = *(const v8bf*)&BsT[col0][half * 8];
    v8bf b0hi = *(const v8bf*)&BsT[col0][16 + half * 8];
    v16bf bf0 = __builtin_shufflevector(b0lo, b0hi,
        0,1,2,3,4,5,6,7,8,9,10,11,12,13,14,15);
    v8bf b1lo = *(const v8bf*)&BsT[col0 + 16][half * 8];
    v8bf b1hi = *(const v8bf*)&BsT[col0 + 16][16 + half * 8];
    v16bf bf1 = __builtin_shufflevector(b1lo, b1hi,
        0,1,2,3,4,5,6,7,8,9,10,11,12,13,14,15);

#pragma unroll
    for (int mt = 0; mt < MT; ++mt) {
      const int row = mt * 16 + l15;
      v8bf alo = *(const v8bf*)&As[row][half * 8];
      v8bf ahi = *(const v8bf*)&As[row][16 + half * 8];
      v16bf afrag = __builtin_shufflevector(alo, ahi,
          0,1,2,3,4,5,6,7,8,9,10,11,12,13,14,15);
      acc[mt][0] = __builtin_amdgcn_wmma_f32_16x16x32_bf16(
          false, afrag, false, bf0, (short)0, acc[mt][0], false, false);
      acc[mt][1] = __builtin_amdgcn_wmma_f32_16x16x32_bf16(
          false, afrag, false, bf1, (short)0, acc[mt][1], false, false);
    }
    __syncthreads();
  }

  // ---- store D ----
#pragma unroll
  for (int s = 0; s < 2; ++s) {
    const int n = nbase + wave * 32 + s * 16 + l15;
    if (n >= N) continue;
#pragma unroll
    for (int mt = 0; mt < MT; ++mt) {
#pragma unroll
      for (int r = 0; r < 8; ++r) {
        int m = mbase + mt * 16 + half * 8 + r;
        if (m < M) cput(batch, m, n, acc[mt][s][r]);
      }
    }
  }
}

// ---------------------------------------------------------------------------
// Attention score GEMM with async global->LDS staging.
// A and B are bf16 in global memory with rows contiguous along K, so every
// LDS tile row is a contiguous 64-byte segment: 4x b128 async loads per row,
// no conversion, no transpose.  Pad rows (m>=M) / cols (n>=N) load garbage
// that only reaches never-stored accumulator lanes.
//   M<=64, N<=128, K multiple of 32.  out[(batch*64+m)*ldn + n].
// ---------------------------------------------------------------------------
template <class AAddr, class BAddr>
__global__ void __launch_bounds__(256)
gemm_attn_async(AAddr aaddr, BAddr baddr, float* out, int ldn,
                int M, int N, int K) {
  __shared__ __align__(16) unsigned short As[64][32];
  __shared__ __align__(16) unsigned short BsT[128][32];

  const int nbase = blockIdx.x * 128;
  const int batch = blockIdx.z;
  const int tid  = threadIdx.x;
  const int wave = tid >> 5;
  const int lane = tid & 31;
  const int half = lane >> 4;
  const int l15  = lane & 15;

  v8f acc[4];
  const v8f vz = {0.f,0.f,0.f,0.f,0.f,0.f,0.f,0.f};
#pragma unroll
  for (int i = 0; i < 4; ++i) acc[i] = vz;

  for (int k0 = 0; k0 < K; k0 += 32) {
    // 64 A rows + 128 B rows, 4 x 16B segments each = 768 transfers
    for (int s = tid; s < 192 * 4; s += 256) {
      int row = s >> 2, seg = s & 3;
      const unsigned short* g;
      unsigned short* l;
      if (row < 64) {
        g = (const unsigned short*)aaddr(batch, row, k0) + seg * 8;
        l = &As[row][seg * 8];
      } else {
        int c = row - 64;
        g = (const unsigned short*)baddr(batch, nbase + c, k0) + seg * 8;
        l = &BsT[c][seg * 8];
      }
#if ATTN_ASYNC
      async_ld16(g, l);
#else
      *(uint4*)l = *(const uint4*)g;
#endif
    }
#if ATTN_ASYNC
    __builtin_amdgcn_s_wait_asynccnt(0);
#endif
    __syncthreads();

    const int col = wave * 16 + l15;
    v8bf blo = *(const v8bf*)&BsT[col][half * 8];
    v8bf bhi = *(const v8bf*)&BsT[col][16 + half * 8];
    v16bf bfrag = __builtin_shufflevector(blo, bhi,
        0,1,2,3,4,5,6,7,8,9,10,11,12,13,14,15);
#pragma unroll
    for (int mt = 0; mt < 4; ++mt) {
      const int row = mt * 16 + l15;
      v8bf alo = *(const v8bf*)&As[row][half * 8];
      v8bf ahi = *(const v8bf*)&As[row][16 + half * 8];
      v16bf afrag = __builtin_shufflevector(alo, ahi,
          0,1,2,3,4,5,6,7,8,9,10,11,12,13,14,15);
      acc[mt] = __builtin_amdgcn_wmma_f32_16x16x32_bf16(
          false, afrag, false, bfrag, (short)0, acc[mt], false, false);
    }
    __syncthreads();
  }

  const int n = nbase + wave * 16 + l15;
  if (n < N) {
#pragma unroll
    for (int mt = 0; mt < 4; ++mt) {
#pragma unroll
      for (int r = 0; r < 8; ++r) {
        int m = mt * 16 + half * 8 + r;
        if (m < M) out[((long)batch * 64 + m) * ldn + n] = acc[mt][r];
      }
    }
  }
}

// Row-address providers for the async attention GEMM (k0 is 32-aligned, and
// 512 % 32 == 0 so the 32 K-values k0..k0+31 stay inside one (d) slice).
struct QAddr {  // q/k bf16 [b][64][50][512]; row index = token p; k = d*512+t
  const unsigned short* q;
  __device__ const void* operator()(int z, int p, int k0) const {
    int b = z >> 3, h = z & 7, d = k0 >> 9, t = k0 & 511;
    return q + ((((long)b * 64 + h * 8 + d) * 50 + p) * 512 + t);
  }
};
struct RelAddr {  // rel_emb bf16 [99][4096]; row index = r
  const unsigned short* r;
  __device__ const void* operator()(int, int n, int k0) const {
    return r + ((long)n * 4096 + k0);
  }
};

// ---------------------------------------------------------------------------
// Functors (im2col views / output scatter)
// ---------------------------------------------------------------------------

// Plain row-major weight matrix A[m][k]
struct AMat {
  const float* p; int lda;
  __device__ float operator()(int, int m, int k) const { return p[(long)m * lda + k]; }
};

// Stem conv im2col.  batch z = b*49 + sp (sp = m7*7+n7).  n = t in [0,1024).
// x logical layout (after the reference transpose) maps to d_in[0] [B,T,3,32,32].
template <int KT, int PAD>
struct LfeB {
  const float* x;
  __device__ float operator()(int z, int k, int n) const {
    int b = z / 49, sp = z % 49, m7 = sp / 7, n7 = sp % 7;
    int tap = k % KT;
    int rem = k / KT;            // ci*64 + kh*8 + kw
    int kw = rem & 7, kh = (rem >> 3) & 7, ci = rem >> 6;
    int tt = n + tap - PAD;
    if (tt < 0 || tt >= 1024) return 0.f;
    int H = m7 * 4 + kh, W = n7 * 4 + kw;
    return x[(((long)b * 1024 + tt) * 3 + ci) * 1024 + H * 32 + W];
  }
};
struct CYConv {  // y[b][64][49][1024]
  float* y; const float* bias; int base;
  __device__ void operator()(int z, int m, int n, float v) const {
    int b = z / 49, sp = z % 49;
    y[(((long)b * 64 + base + m) * 49 + sp) * 1024 + n] = v + bias[m];
  }
};

// Token-layout activation B[k=c][n=(p,t)]  (src is [b][64][50][512] f32)
struct TokB {
  const float* z;
  __device__ float operator()(int b, int k, int n) const {
    int p = n >> 9, t = n & 511;
    return z[(((long)b * 64 + k) * 50 + p) * 512 + t];
  }
};
struct CQkv {  // q,k stored bf16 (GEMM-only consumers); v stays f32
  unsigned short *q, *k; float* v; const float* bias;
  __device__ void operator()(int b, int m, int n, float val) const {
    val += bias[m];
    int p = n >> 9, t = n & 511, c = m & 63;
    long idx = (((long)b * 64 + c) * 50 + p) * 512 + t;
    if (m < 64)       q[idx] = f2bfbits(val);
    else if (m < 128) k[idx] = f2bfbits(val);
    else              v[idx] = val;
  }
};

struct AttnA {  // attn [z][64][64] f32 (rows/cols >= 50 zero / unread; K=50)
  const float* at;
  __device__ float operator()(int z, int m, int k) const {
    return at[((long)z * 64 + m) * 64 + k];
  }
};
struct VB {  // B[k=p'][n=dt]
  const float* v;
  __device__ float operator()(int z, int k, int n) const {
    int b = z >> 3, h = z & 7, d = n >> 9, t = n & 511;
    return v[(((long)b * 64 + h * 8 + d) * 50 + k) * 512 + t];
  }
};
struct CAvRes {  // out (p, dt) -> mha[b][c][p][t]  (+ residual fused)
  float* o; const float* res;
  __device__ void operator()(int z, int m, int n, float val) const {
    int b = z >> 3, h = z & 7, d = n >> 9, t = n & 511;
    long idx = (((long)b * 64 + h * 8 + d) * 50 + m) * 512 + t;
    o[idx] = val + res[idx];
  }
};

// CFE 1-D conv over t (kernel KT, pad PAD):  k = c*KT + tap
template <int KT, int PAD>
struct CfeB {
  const float* z;
  __device__ float operator()(int b, int k, int n) const {
    int p = n >> 9, t = n & 511;
    int c = k / KT, tap = k % KT, tt = t + tap - PAD;
    if (tt < 0 || tt >= 512) return 0.f;
    return z[(((long)b * 64 + c) * 50 + p) * 512 + tt];
  }
};
struct CTok {
  float* dst; const float* bias; int base;
  __device__ void operator()(int b, int m, int n, float v) const {
    int p = n >> 9, t = n & 511;
    dst[(((long)b * 64 + base + m) * 50 + p) * 512 + t] = v + bias[m];
  }
};
struct CTokRes {  // c3 output + bias + residual fused
  float* dst; const float* bias; const float* res;
  __device__ void operator()(int b, int m, int n, float v) const {
    int p = n >> 9, t = n & 511;
    long idx = (((long)b * 64 + m) * 50 + p) * 512 + t;
    dst[idx] = v + bias[m] + res[idx];
  }
};

// Head conv: k = (c*49 + j)*KT + tap, j = token rows 0..48, conv only over t
template <int KT, int PAD>
struct HeadB {
  const float* z;
  __device__ float operator()(int b, int k, int n) const {
    int tap = k % KT, rem = k / KT, j = rem % 49, c = rem / 49;
    int tt = n + tap - PAD;
    if (tt < 0 || tt >= 512) return 0.f;
    return z[(((long)b * 64 + c) * 50 + j) * 512 + tt];
  }
};
struct CHead {
  float* h; const float* bias; int base;
  __device__ void operator()(int b, int m, int n, float v) const {
    h[((long)b * 64 + base + m) * 512 + n] = v + bias[m];
  }
};

// FC1: A = l1_w [128][16384], B[k][n=b] = feat[b][k]
struct FeatB {
  const float* f;
  __device__ float operator()(int, int k, int n) const { return f[(long)n * 16384 + k]; }
};
struct CL1 {
  float* o; const float* bias;
  __device__ void operator()(int, int m, int n, float v) const {
    o[(long)n * 128 + m] = v + bias[m];
  }
};

// ---------------------------------------------------------------------------
// Pointwise / reduction kernels
// ---------------------------------------------------------------------------
__global__ void k_stats(const float* __restrict__ x, float* mean, float* var,
                        int C, int SP, int B) {
  int c = blockIdx.x;
  float s = 0.f, s2 = 0.f;
  long n = (long)B * SP;
  for (long e = threadIdx.x; e < n; e += 256) {
    int b = (int)(e / SP), idx = (int)(e % SP);
    float v = x[((long)b * C + c) * SP + idx];
    s += v; s2 += v * v;
  }
  __shared__ float rs[256], rq[256];
  rs[threadIdx.x] = s; rq[threadIdx.x] = s2;
  __syncthreads();
  for (int st = 128; st > 0; st >>= 1) {
    if ((int)threadIdx.x < st) { rs[threadIdx.x] += rs[threadIdx.x + st]; rq[threadIdx.x] += rq[threadIdx.x + st]; }
    __syncthreads();
  }
  if (threadIdx.x == 0) {
    float m = rs[0] / (float)n;
    mean[c] = m;
    var[c] = rq[0] / (float)n - m * m;   // biased variance (jnp.var)
  }
}

__global__ void k_bn_apply(const float* __restrict__ src, float* __restrict__ dst,
                           const float* g, const float* be,
                           const float* mean, const float* var,
                           int C, int SP, long total, int elu) {
  long i = (long)blockIdx.x * 256 + threadIdx.x;
  if (i >= total) return;
  int c = (int)((i / SP) % C);
  float v = (src[i] - mean[c]) * rsqrtf(var[c] + 1e-5f) * g[c] + be[c];
  if (elu) v = v > 0.f ? v : (__expf(v) - 1.f);
  dst[i] = v;
}

__global__ void k_cvt_bf(const float* __restrict__ src, unsigned short* __restrict__ dst,
                         long total) {
  long i = (long)blockIdx.x * 256 + threadIdx.x;
  if (i >= total) return;
  dst[i] = f2bfbits(src[i]);
}

// BN+ELU+maxpool(t,2) on stem output -> spatial tokens 1..49
__global__ void k_lfe_post(const float* __restrict__ y, float* __restrict__ z,
                           const float* g, const float* be,
                           const float* mean, const float* var) {
  long i = (long)blockIdx.x * 256 + threadIdx.x;
  if (i >= 16L * 64 * 49 * 512) return;
  int tp = i & 511; long r = i >> 9;
  int sp = (int)(r % 49); r /= 49;
  int c = (int)(r % 64);  int b = (int)(r / 64);
  const float* s = y + (((long)b * 64 + c) * 49 + sp) * 1024 + 2 * tp;
  float sc = rsqrtf(var[c] + 1e-5f) * g[c], mb = mean[c], bt = be[c];
  float a0 = (s[0] - mb) * sc + bt; a0 = a0 > 0.f ? a0 : (__expf(a0) - 1.f);
  float a1 = (s[1] - mb) * sc + bt; a1 = a1 > 0.f ? a1 : (__expf(a1) - 1.f);
  z[(((long)b * 64 + c) * 50 + 1 + sp) * 512 + tp] = fmaxf(a0, a1);
}

// channel token = eye(64,512) parameter, broadcast over batch
__global__ void k_fill_ct(float* __restrict__ z, const float* __restrict__ ct) {
  long i = (long)blockIdx.x * 256 + threadIdx.x;
  if (i >= 16L * 64 * 512) return;
  int t = i & 511; int c = (int)((i >> 9) & 63); int b = (int)(i >> 15);
  z[(((long)b * 64 + c) * 50 + 0) * 512 + t] = ct[c * 512 + t];
}

// softmax((dots + rel_to_abs(rel)) * scale); rows/cols padded to 64 with zeros
__global__ void k_softmax(const float* __restrict__ dots, const float* __restrict__ rels,
                          float* __restrict__ attn, float scale) {
  int row = blockIdx.x & 63;
  int z = blockIdx.x >> 6;
  int col = threadIdx.x;
  float* orow = attn + ((long)z * 64 + row) * 64;
  if (row >= 50) { orow[col] = 0.f; return; }
  float v = -3.0e38f;
  if (col < 50) {
    float d = dots[((long)z * 64 + row) * 64 + col];
    float r = rels[((long)z * 64 + row) * 112 + 49 + col - row];  // rel_to_abs
    v = (d + r) * scale;
  }
  __shared__ float red[64];
  red[col] = v; __syncthreads();
  for (int s = 32; s > 0; s >>= 1) { if (col < s) red[col] = fmaxf(red[col], red[col + s]); __syncthreads(); }
  float mx = red[0]; __syncthreads();
  float e = (col < 50) ? __expf(v - mx) : 0.f;
  red[col] = e; __syncthreads();
  for (int s = 32; s > 0; s >>= 1) { if (col < s) red[col] += red[col + s]; __syncthreads(); }
  orow[col] = e / red[0];
}

// BN3d(1 feature) + ELU + maxpool(2) -> flattened features
__global__ void k_head_post(const float* __restrict__ head, float* __restrict__ feat,
                            const float* hg, const float* hb,
                            const float* mean, const float* var) {
  long i = (long)blockIdx.x * 256 + threadIdx.x;
  if (i >= 16L * 64 * 256) return;
  int tp = i & 255; int c = (int)((i >> 8) & 63); int b = (int)(i >> 14);
  const float* s = head + ((long)b * 64 + c) * 512 + 2 * tp;
  float sc = rsqrtf(var[0] + 1e-5f) * hg[0], mb = mean[0], bt = hb[0];
  float a0 = (s[0] - mb) * sc + bt; a0 = a0 > 0.f ? a0 : (__expf(a0) - 1.f);
  float a1 = (s[1] - mb) * sc + bt; a1 = a1 > 0.f ? a1 : (__expf(a1) - 1.f);
  feat[(long)b * 16384 + c * 256 + tp] = fmaxf(a0, a1);
}

__global__ void k_l2(const float* __restrict__ l1o, const float* __restrict__ w,
                     const float* __restrict__ bias, float* __restrict__ out) {
  int b = blockIdx.x, c = threadIdx.x;
  if (c >= 40) return;
  float s = bias[c];
  for (int j = 0; j < 128; ++j) s += l1o[b * 128 + j] * w[c * 128 + j];
  out[b * 40 + c] = s;
}

// ---------------------------------------------------------------------------
// Host-side orchestration
// ---------------------------------------------------------------------------
extern "C" void kernel_launch(void* const* d_in, const int* in_sizes, int n_in,
                              void* d_out, int out_size, void* d_ws, size_t ws_size,
                              hipStream_t stream) {
  (void)in_sizes; (void)n_in; (void)out_size; (void)ws_size;

  // ---- input mapping: x first, then params flattened in jax pytree order ----
  const float* X = (const float*)d_in[0];
  struct BlkP {
    const float *bn1_b, *bn1_g, *bn2_b, *bn2_g, *c1_b, *c1_w, *c2_b, *c2_w,
                *c3_b, *c3_w, *cbn_b, *cbn_g, *qkv_b, *qkv_w, *rel_emb;
  };
  int ii = 1;
  BlkP blk[2];
  for (int bI = 0; bI < 2; ++bI) {
    blk[bI].bn1_b  = (const float*)d_in[ii++];
    blk[bI].bn1_g  = (const float*)d_in[ii++];
    blk[bI].bn2_b  = (const float*)d_in[ii++];
    blk[bI].bn2_g  = (const float*)d_in[ii++];
    blk[bI].c1_b   = (const float*)d_in[ii++];
    blk[bI].c1_w   = (const float*)d_in[ii++];
    blk[bI].c2_b   = (const float*)d_in[ii++];
    blk[bI].c2_w   = (const float*)d_in[ii++];
    blk[bI].c3_b   = (const float*)d_in[ii++];
    blk[bI].c3_w   = (const float*)d_in[ii++];
    blk[bI].cbn_b  = (const float*)d_in[ii++];
    blk[bI].cbn_g  = (const float*)d_in[ii++];
    blk[bI].qkv_b  = (const float*)d_in[ii++];
    blk[bI].qkv_w  = (const float*)d_in[ii++];
    blk[bI].rel_emb= (const float*)d_in[ii++];
  }
  const float* channel_token = (const float*)d_in[ii++];
  const float* h1_b  = (const float*)d_in[ii++];
  const float* h1_w  = (const float*)d_in[ii++];
  const float* h2_b  = (const float*)d_in[ii++];
  const float* h2_w  = (const float*)d_in[ii++];
  const float* hbn_b = (const float*)d_in[ii++];
  const float* hbn_g = (const float*)d_in[ii++];
  const float* l1_b  = (const float*)d_in[ii++];
  const float* l1_w  = (const float*)d_in[ii++];
  const float* l2_b  = (const float*)d_in[ii++];
  const float* l2_w  = (const float*)d_in[ii++];
  const float* lfe_b3 = (const float*)d_in[ii++];
  const float* lfe_b5 = (const float*)d_in[ii++];
  const float* lfe_bt = (const float*)d_in[ii++];
  const float* lfe_g  = (const float*)d_in[ii++];
  const float* lfe_w3 = (const float*)d_in[ii++];
  const float* lfe_w5 = (const float*)d_in[ii++];

  // ---- workspace carve ----
  char* wsb = (char*)d_ws;
  size_t off = 0;
  auto alloc = [&](size_t bytes) -> void* {
    void* p = (void*)(wsb + off);
    off += (bytes + 255) & ~(size_t)255;
    return p;
  };
  const long TOK_ELEMS = 16L * 64 * 50 * 512;              // 26.2M
  float* yconv = (float*)alloc(16L * 64 * 49 * 1024 * 4);  // stem conv raw
  float* ztok  = (float*)alloc(TOK_ELEMS * 4);             // running x
  unsigned short* qb = (unsigned short*)alloc(TOK_ELEMS * 2);   // bf16
  unsigned short* kb = (unsigned short*)alloc(TOK_ELEMS * 2);   // bf16
  float* vb    = (float*)alloc(TOK_ELEMS * 4);
  float* mha   = (float*)alloc(TOK_ELEMS * 4);             // mha / c3 (+res)
  float* ffd   = (float*)alloc(TOK_ELEMS * 4);
  float* dots  = (float*)alloc(16L * 8 * 64 * 64 * 4);
  float* rels  = (float*)alloc(16L * 8 * 64 * 112 * 4);
  float* attn  = (float*)alloc(16L * 8 * 64 * 64 * 4);
  unsigned short* rel_bf = (unsigned short*)alloc(99L * 4096 * 2);
  float* head  = (float*)alloc(16L * 64 * 512 * 4);
  float* feat  = (float*)alloc(16L * 16384 * 4);
  float* l1o   = (float*)alloc(16L * 128 * 4);
  float* stat_m = (float*)alloc(512 * 4);
  float* stat_v = (float*)alloc(512 * 4);
  float* outp = (float*)d_out;

  auto blocks = [](long n) { return (unsigned)((n + 255) / 256); };

  // ---- stem: two 3-D convs as implicit GEMM (M=32 -> MT=2, no padded rows) ----
  gemm_bf16<2><<<dim3(4, 1, 16 * 49), 256, 0, stream>>>(
      AMat{lfe_w3, 576}, LfeB<3, 1>{X}, CYConv{yconv, lfe_b3, 0}, 32, 1024, 576);
  gemm_bf16<2><<<dim3(4, 1, 16 * 49), 256, 0, stream>>>(
      AMat{lfe_w5, 960}, LfeB<5, 2>{X}, CYConv{yconv, lfe_b5, 32}, 32, 1024, 960);
  k_stats<<<64, 256, 0, stream>>>(yconv, stat_m, stat_v, 64, 49 * 1024, 16);
  k_lfe_post<<<blocks(16L * 64 * 49 * 512), 256, 0, stream>>>(
      yconv, ztok, lfe_g, lfe_bt, stat_m, stat_v);
  k_fill_ct<<<blocks(16L * 64 * 512), 256, 0, stream>>>(ztok, channel_token);

  // ---- transformer blocks ----
  const float scale = 0.35355339059327373f;  // (C/H)^-0.5 = 8^-0.5
  for (int bi = 0; bi < 2; ++bi) {
    const BlkP& P = blk[bi];
    // qkv 1x1 conv:  M=192, K=64, N=50*512 per batch  (q,k -> bf16; v -> f32)
    gemm_bf16<4><<<dim3(100, 3, 16), 256, 0, stream>>>(
        AMat{P.qkv_w, 64}, TokB{ztok}, CQkv{qb, kb, vb, P.qkv_b}, 192, 25600, 64);
    // rel_emb -> bf16 (per block)
    k_cvt_bf<<<blocks(99L * 4096), 256, 0, stream>>>(P.rel_emb, rel_bf, 99L * 4096);
    // Q K^T : per (b,h)  M=50, N=50, K=4096   (async LDS staging)
    gemm_attn_async<<<dim3(1, 1, 128), 256, 0, stream>>>(
        QAddr{qb}, QAddr{kb}, dots, 64, 50, 50, 4096);
    // Q rel^T : M=50, N=99, K=4096
    gemm_attn_async<<<dim3(1, 1, 128), 256, 0, stream>>>(
        QAddr{qb}, RelAddr{rel_bf}, rels, 112, 50, 99, 4096);
    k_softmax<<<16 * 8 * 64, 64, 0, stream>>>(dots, rels, attn, scale);
    // attn V : M=50, N=4096, K=50  (residual add fused into output)
    gemm_bf16<4><<<dim3(16, 1, 128), 256, 0, stream>>>(
        AttnA{attn}, VB{vb}, CAvRes{mha, ztok}, 50, 4096, 50);
    // BN1
    k_stats<<<64, 256, 0, stream>>>(mha, stat_m, stat_v, 64, 25600, 16);
    k_bn_apply<<<blocks(TOK_ELEMS), 256, 0, stream>>>(
        mha, ztok, P.bn1_g, P.bn1_b, stat_m, stat_v, 64, 25600, TOK_ELEMS, 0);
    // CFE conv branches (M=32 -> MT=2)
    gemm_bf16<2><<<dim3(100, 1, 16), 256, 0, stream>>>(
        AMat{P.c1_w, 192}, CfeB<3, 1>{ztok}, CTok{ffd, P.c1_b, 0}, 32, 25600, 192);
    gemm_bf16<2><<<dim3(100, 1, 16), 256, 0, stream>>>(
        AMat{P.c2_w, 320}, CfeB<5, 2>{ztok}, CTok{ffd, P.c2_b, 32}, 32, 25600, 320);
    k_stats<<<64, 256, 0, stream>>>(ffd, stat_m, stat_v, 64, 25600, 16);
    k_bn_apply<<<blocks(TOK_ELEMS), 256, 0, stream>>>(
        ffd, ffd, P.cbn_g, P.cbn_b, stat_m, stat_v, 64, 25600, TOK_ELEMS, 1);
    // c3 1x1 (+bias+residual fused) + BN2
    gemm_bf16<4><<<dim3(100, 1, 16), 256, 0, stream>>>(
        AMat{P.c3_w, 64}, TokB{ffd}, CTokRes{mha, P.c3_b, ztok}, 64, 25600, 64);
    k_stats<<<64, 256, 0, stream>>>(mha, stat_m, stat_v, 64, 25600, 16);
    k_bn_apply<<<blocks(TOK_ELEMS), 256, 0, stream>>>(
        mha, ztok, P.bn2_g, P.bn2_b, stat_m, stat_v, 64, 25600, TOK_ELEMS, 0);
  }

  // ---- head convs (stride 49 over tokens => one vertical position) ----
  gemm_bf16<2><<<dim3(2, 1, 16), 256, 0, stream>>>(
      AMat{h1_w, 9408}, HeadB<3, 1>{ztok}, CHead{head, h1_b, 0}, 32, 512, 9408);
  gemm_bf16<2><<<dim3(2, 1, 16), 256, 0, stream>>>(
      AMat{h2_w, 15680}, HeadB<5, 2>{ztok}, CHead{head, h2_b, 32}, 32, 512, 15680);
  // BN3d with a single feature: global stats over everything
  k_stats<<<1, 256, 0, stream>>>(head, stat_m, stat_v, 1, 64 * 512, 16);
  k_head_post<<<blocks(16L * 64 * 256), 256, 0, stream>>>(
      head, feat, hbn_g, hbn_b, stat_m, stat_v);

  // ---- classifier ----
  gemm_bf16<4><<<dim3(1, 2, 1), 256, 0, stream>>>(
      AMat{l1_w, 16384}, FeatB{feat}, CL1{l1o, l1_b}, 128, 16, 16384);
  k_l2<<<16, 64, 0, stream>>>(l1o, l2_w, l2_b, outp);
}
